// GPTLanguageModel_86595130622142
// MI455X (gfx1250) — matrix operand order repaired and checked by reading, not compiled, not run on hardware
//
#include <hip/hip_runtime.h>
#include <hip/hip_bf16.h>

typedef __attribute__((ext_vector_type(16))) _Float16 v16h;
typedef __attribute__((ext_vector_type(8)))  _Float16 v8h;
typedef __attribute__((ext_vector_type(4)))  _Float16 v4h;
typedef __attribute__((ext_vector_type(8)))  float    v8f;
typedef __attribute__((ext_vector_type(4)))  float    v4f;

#define DD   1024
#define TT   1024
#define BBA  4
#define HH   16
#define HDIM 64
#define VV   32000
#define NROW (BBA * TT)   // 4096
#define FFD  4096         // 4*D

// Tensor Data Mover path (guarded; this toolchain uses the 6-arg builtin).
#if defined(__has_builtin)
#if __has_builtin(__builtin_amdgcn_tensor_load_to_lds) && \
    __has_builtin(__builtin_amdgcn_s_wait_tensorcnt)
#define GPT_TDM 1
#endif
#endif

#ifdef GPT_TDM
typedef __attribute__((ext_vector_type(4))) unsigned int u32x4;
typedef __attribute__((ext_vector_type(8))) int          i32x8;
typedef __attribute__((ext_vector_type(4))) int          i32x4;
#endif

__device__ __forceinline__ v8f wmma_f16(v16h a, v16h b, v8f c) {
  // D = A(16x32 f16) * B(32x16 f16) + C(16x16 f32)
  return __builtin_amdgcn_wmma_f32_16x16x32_f16(
      /*neg_a=*/false, a, /*neg_b=*/false, b,
      /*c_mod=*/(short)0, c, /*reuse_a=*/false, /*reuse_b=*/false);
}

// ---------------------------------------------------------------------------
// x[row, d] = tok_emb[idx[row], d] + pos_emb[row % T, d]
__global__ void k_embed(const int* __restrict__ idx, const float* __restrict__ tok,
                        const float* __restrict__ pos, float* __restrict__ x) {
  int row = blockIdx.x;
  int t = row & (TT - 1);
  int tokid = idx[row];
  for (int d = threadIdx.x; d < DD; d += blockDim.x)
    x[(size_t)row * DD + d] = tok[(size_t)tokid * DD + d] + pos[(size_t)t * DD + d];
}

// ---------------------------------------------------------------------------
// Row layernorm over D=1024, f32 in -> f16 out.  One 256-thread block per row.
__global__ __launch_bounds__(256) void k_layernorm(const float* __restrict__ x,
                                                   const float* __restrict__ g,
                                                   const float* __restrict__ b,
                                                   _Float16* __restrict__ y) {
  __shared__ float s1[8], s2[8];
  int row = blockIdx.x;
  const float* xr = x + (size_t)row * DD;
  int lane = threadIdx.x & 31, wv = threadIdx.x >> 5;
  float v[4], s = 0.f, ss = 0.f;
#pragma unroll
  for (int i = 0; i < 4; i++) {
    v[i] = xr[threadIdx.x + 256 * i];
    s += v[i]; ss += v[i] * v[i];
  }
#pragma unroll
  for (int off = 16; off; off >>= 1) {
    s  += __shfl_down(s,  off, 32);
    ss += __shfl_down(ss, off, 32);
  }
  if (!lane) { s1[wv] = s; s2[wv] = ss; }
  __syncthreads();
  if (threadIdx.x == 0) {
    float a = 0.f, c = 0.f;
    for (int i = 0; i < 8; i++) { a += s1[i]; c += s2[i]; }
    s1[0] = a; s2[0] = c;
  }
  __syncthreads();
  float mu  = s1[0] * (1.0f / DD);
  float var = s2[0] * (1.0f / DD) - mu * mu;
  float inv = rsqrtf(var + 1e-5f);
#pragma unroll
  for (int i = 0; i < 4; i++) {
    int d = threadIdx.x + 256 * i;
    y[(size_t)row * DD + d] = (_Float16)((v[i] - mu) * inv * g[d] + b[d]);
  }
}

// ---------------------------------------------------------------------------
// Streaming f32 -> f16 conversion, 4 elements per thread (b128 in / b64 out),
// non-temporal so one-shot weight streams don't evict the L2 working set.
__global__ void k_f32_to_f16_v4(const float* __restrict__ in,
                                _Float16* __restrict__ out, long long n4) {
  long long i = (long long)blockIdx.x * 256 + threadIdx.x;
  if (i < n4) {
    v4f a = __builtin_nontemporal_load((const v4f*)in + i);
    v4h b;
#pragma unroll
    for (int j = 0; j < 4; j++) b[j] = (_Float16)a[j];
    __builtin_nontemporal_store(b, (v4h*)out + i);
  }
}

// wq/wk/wv: (H, D, HD) f32 -> (D, H*HD) f16 so head h occupies cols [h*64, h*64+64)
__global__ void k_repack_hde(const float* __restrict__ w, _Float16* __restrict__ out) {
  int i = blockIdx.x * 256 + threadIdx.x;   // over D*D/4, 4 contiguous e per thread
  int o = i << 2;
  int d = o >> 10;
  int c = o & 1023;
  int h = c >> 6, e = c & 63;               // e multiple of 4
  v4f a = __builtin_nontemporal_load(
      (const v4f*)(w + ((size_t)((h << 10) + d)) * 64 + e));
  v4h b;
#pragma unroll
  for (int j = 0; j < 4; j++) b[j] = (_Float16)a[j];
  *(v4h*)(out + o) = b;
}

// ---------------------------------------------------------------------------
// C[M,N] = A[M,K](f16) * B[K,N](f16)  (+bias)(+residual)(+relu) -> f32 / f16
// Block tile 128x128, BK=32, 8 waves (4x2), wave tile 32x64 (2x4 WMMA tiles).
// A tile staged by the Tensor Data Mover (pad_enable gives the 40-half pitch);
// B tile staged transposed by the waves.  Epilogue options are template params.
#define BM 128
#define BN 128
#define BK 32
template <bool HAS_BIAS, bool HAS_RES, bool RELU, bool OUTF, bool OUTH>
__global__ __launch_bounds__(256) void k_gemm(
    const _Float16* __restrict__ A, const _Float16* __restrict__ B,
    const float* __restrict__ bias, const float* __restrict__ resid,
    float* __restrict__ Cf, _Float16* __restrict__ Ch,
    int M, int N, int K) {
  __shared__ _Float16 As[BM][BK + 8];
  __shared__ _Float16 Bt[BN][BK + 8];   // Bt[n][k] = B[k][n]

  int tid  = threadIdx.x;
  int wave = tid >> 5, lane = tid & 31;
  int waveM = wave >> 1;           // 0..3
  int waveN = wave & 1;            // 0..1
  int m0 = blockIdx.y * BM, n0 = blockIdx.x * BN;
  int lm = lane & 15, kg = lane >> 4;

  v8f acc[2][4] = {};

  int arow = tid >> 1, acol = (tid & 1) * 16;     // A stage (fallback)
  int bkr  = tid >> 3, bnb  = (tid & 7) * 16;     // B stage: 32 k-rows x 128

#ifdef GPT_TDM
  // Constant pieces of the A-tile tensor descriptor (D#).
  unsigned lds_a = (unsigned)(uintptr_t)&As[0][0];  // flat LDS addr low 32b = offset
#endif

  for (int k0 = 0; k0 < K; k0 += BK) {
    __syncthreads();
#ifdef GPT_TDM
    if (wave == 0) {
      unsigned long long ga =
          (unsigned long long)(uintptr_t)(A + (size_t)m0 * K + k0);
      u32x4 g0 = {};
      i32x8 g1 = {};
      i32x4 g2 = {}, g3 = {};
      i32x8 g4 = {};
      g0[0] = 1u;                                   // count=1 (valid user D#)
      g0[1] = lds_a;                                // lds_addr (bytes)
      g0[2] = (unsigned)ga;                         // global_addr[31:0]
      g0[3] = (unsigned)(ga >> 32) | (2u << 30);    // global_addr[56:32] | type=2
      g1[0] = (1 << 16)                             // data_size = 2 bytes
            | (1 << 20)                             // pad_enable
            | (3 << 22)                             // pad_interval = 16 DWORDs
            | (3 << 25);                            // pad_amount = 4 DWORDs (pitch 40h)
      g1[1] = (BK & 0xFFFF) << 16;                  // tensor_dim0 = 32
      g1[2] = (BM & 0xFFFF) << 16;                  // tensor_dim1 = 128
      g1[3] = (BK & 0xFFFF) << 16;                  // tile_dim0 = 32
      g1[4] = BM;                                   // tile_dim1 = 128, tile_dim2 = 0
      g1[5] = K;                                    // tensor_dim0_stride = K
      g1[6] = 0;
      g1[7] = 0;
      __builtin_amdgcn_tensor_load_to_lds(g0, g1, g2, g3, g4, 0);
      __builtin_amdgcn_s_wait_tensorcnt(0);
    }
#else
    {
      const v8h* src = (const v8h*)(A + (size_t)(m0 + arow) * K + k0 + acol);
      *(v8h*)&As[arow][acol]     = src[0];
      *(v8h*)&As[arow][acol + 8] = src[1];
    }
#endif
    {
      const _Float16* src = B + (size_t)(k0 + bkr) * N + n0 + bnb;
#pragma unroll
      for (int j = 0; j < 16; j++) Bt[bnb + j][bkr] = __builtin_nontemporal_load(src + j);
    }
    // prefetch next B K-tile behind the WMMA burst (global_prefetch_b8)
    if (k0 + BK < K) {
      __builtin_prefetch(B + (size_t)(k0 + BK + bkr) * N + n0 + bnb, 0, 0);
    }
    __syncthreads();

    v16h af[2];
#pragma unroll
    for (int ti = 0; ti < 2; ti++) {
      int m = waveM * 32 + ti * 16 + lm;
      const v8h* p0 = (const v8h*)&As[m][kg * 16];       // K = kg*16 .. +7
      const v8h* p1 = (const v8h*)&As[m][kg * 16 + 8];   // K = kg*16+8 .. +15
#pragma unroll
      for (int e = 0; e < 8; e++) { af[ti][e] = (*p0)[e]; af[ti][e + 8] = (*p1)[e]; }
    }
    v16h bf[4];
#pragma unroll
    for (int tj = 0; tj < 4; tj++) {
      int n = waveN * 64 + tj * 16 + lm;
      const v8h* q0 = (const v8h*)&Bt[n][kg * 16];       // K = kg*16 .. +7
      const v8h* q1 = (const v8h*)&Bt[n][kg * 16 + 8];   // K = kg*16+8 .. +15
#pragma unroll
      for (int e = 0; e < 8; e++) { bf[tj][e] = (*q0)[e]; bf[tj][e + 8] = (*q1)[e]; }
    }
#pragma unroll
    for (int ti = 0; ti < 2; ti++)
#pragma unroll
      for (int tj = 0; tj < 4; tj++)
        acc[ti][tj] = wmma_f16(af[ti], bf[tj], acc[ti][tj]);
  }

  // Branch-free epilogue: lane holds (row = base + 8*kg + r, col = base + lm).
#pragma unroll
  for (int tj = 0; tj < 4; tj++) {
    int colg = n0 + waveN * 64 + tj * 16 + lm;
    float bv = 0.f;
    if constexpr (HAS_BIAS) bv = bias[colg];
#pragma unroll
    for (int ti = 0; ti < 2; ti++) {
#pragma unroll
      for (int r = 0; r < 8; r++) {
        int rowg = m0 + waveM * 32 + ti * 16 + kg * 8 + r;
        size_t off = (size_t)rowg * N + colg;
        float val = acc[ti][tj][r] + bv;
        if constexpr (HAS_RES) val += resid[off];
        if constexpr (RELU)    val = fmaxf(val, 0.f);
        if constexpr (OUTF)    Cf[off] = val;
        if constexpr (OUTH)    Ch[off] = (_Float16)val;
      }
    }
  }
}

// ---------------------------------------------------------------------------
// Attention: one block = 32 query rows of one (b, h). Scores (32x1024 f32),
// softmaxed P (f16) and a transposed V staging tile all live in LDS (~202KB of
// the 320KB WGP budget). Q*K^T and P*V both via WMMA.
#define QT 32
#define ATT_LDS (QT * TT * 4 + QT * TT * 2 + 64 * 40 * 2)
__global__ __launch_bounds__(256) void k_attn(const _Float16* __restrict__ q,
                                              const _Float16* __restrict__ k,
                                              const _Float16* __restrict__ v,
                                              _Float16* __restrict__ o) {
  extern __shared__ char smem[];
  float    (*S)[TT]  = (float(*)[TT])smem;                               // 128 KB
  _Float16 (*P)[TT]  = (_Float16(*)[TT])(smem + QT * TT * 4);            //  64 KB
  _Float16 (*Vt)[40] = (_Float16(*)[40])(smem + QT * TT * 4 + QT * TT * 2);

  int qt = blockIdx.x, h = blockIdx.y, b = blockIdx.z;
  int tid = threadIdx.x, wave = tid >> 5, lane = tid & 31;
  int lm = lane & 15, kg = lane >> 4;

  const _Float16* qbase = q + ((size_t)(b * TT + qt * QT)) * DD + h * HDIM;
  const _Float16* kbase = k + ((size_t)(b * TT)) * DD + h * HDIM;
  const _Float16* vbase = v + ((size_t)(b * TT)) * DD + h * HDIM;
  const float scale = 0.03125f;  // D^-0.5 = 1024^-0.5 (reference scales by n_embd)

  // ---- Phase 1: S = scale * Q K^T, causal mask.  2 x 64 tiles, 16 per wave.
  for (int t = wave; t < 128; t += 8) {
    int mt = t & 1, nt = t >> 1;
    v8f acc = {};
#pragma unroll
    for (int kk = 0; kk < HDIM; kk += 32) {
      v16h af, bf;
      {
        int m = mt * 16 + lm;
        const v8h* p0 = (const v8h*)(qbase + (size_t)m * DD + kk + kg * 16);
        const v8h* p1 = (const v8h*)(qbase + (size_t)m * DD + kk + kg * 16 + 8);
#pragma unroll
        for (int e = 0; e < 8; e++) { af[e] = (*p0)[e]; af[e + 8] = (*p1)[e]; }
      }
      {
        int s = nt * 16 + lm;  // key index; K rows are naturally B^T layout
        const v8h* p0 = (const v8h*)(kbase + (size_t)s * DD + kk + kg * 16);
        const v8h* p1 = (const v8h*)(kbase + (size_t)s * DD + kk + kg * 16 + 8);
#pragma unroll
        for (int e = 0; e < 8; e++) { bf[e] = (*p0)[e]; bf[e + 8] = (*p1)[e]; }
      }
      acc = wmma_f16(af, bf, acc);
    }
    int col = nt * 16 + lm;
#pragma unroll
    for (int r = 0; r < 8; r++) {
      int row = mt * 16 + kg * 8 + r;
      int gq = qt * QT + row;
      float val = acc[r] * scale;
      if (col > gq) val = -1e30f;   // causal mask
      S[row][col] = val;
    }
  }
  __syncthreads();

  // ---- Phase 2: row softmax -> P (f16). 4 rows per wave, 32 cols per lane.
  for (int r = wave * 4; r < wave * 4 + 4; r++) {
    float vv[32];
    float mx = -1e30f;
#pragma unroll
    for (int j = 0; j < 32; j++) { vv[j] = S[r][lane + 32 * j]; mx = fmaxf(mx, vv[j]); }
#pragma unroll
    for (int off = 16; off; off >>= 1) mx = fmaxf(mx, __shfl_xor(mx, off, 32));
    float sum = 0.f;
#pragma unroll
    for (int j = 0; j < 32; j++) { vv[j] = __expf(vv[j] - mx); sum += vv[j]; }
#pragma unroll
    for (int off = 16; off; off >>= 1) sum += __shfl_xor(sum, off, 32);
    float inv = 1.f / sum;
#pragma unroll
    for (int j = 0; j < 32; j++) P[r][lane + 32 * j] = (_Float16)(vv[j] * inv);
  }
  __syncthreads();

  // ---- Phase 3: O = P * V.  Output 32x64 = 8 WMMA tiles, one per wave.
  v8f acc = {};
  int mt = wave & 1, nt = wave >> 1;
  for (int ks = 0; ks < TT; ks += 32) {
    __syncthreads();
    {  // stage Vt[n=headdim][k=key] = V[ks+k][n]
      int krow = tid >> 3;
      int nb = (tid & 7) * 8;
      const _Float16* src = vbase + (size_t)(ks + krow) * DD + nb;
#pragma unroll
      for (int j = 0; j < 8; j++) Vt[nb + j][krow] = src[j];
    }
    __syncthreads();
    v16h af, bf;
    {
      int m = mt * 16 + lm;
      const v8h* p0 = (const v8h*)&P[m][ks + kg * 16];
      const v8h* p1 = (const v8h*)&P[m][ks + kg * 16 + 8];
#pragma unroll
      for (int e = 0; e < 8; e++) { af[e] = (*p0)[e]; af[e + 8] = (*p1)[e]; }
    }
    {
      int n = nt * 16 + lm;
      const v8h* p0 = (const v8h*)&Vt[n][kg * 16];
      const v8h* p1 = (const v8h*)&Vt[n][kg * 16 + 8];
#pragma unroll
      for (int e = 0; e < 8; e++) { bf[e] = (*p0)[e]; bf[e + 8] = (*p1)[e]; }
    }
    acc = wmma_f16(af, bf, acc);
  }
  {
    int col = nt * 16 + lm;
    _Float16* obase = o + ((size_t)(b * TT + qt * QT)) * DD + h * HDIM;
#pragma unroll
    for (int r = 0; r < 8; r++) {
      int row = mt * 16 + kg * 8 + r;
      obase[(size_t)row * DD + col] = (_Float16)acc[r];
    }
  }
}

// ---------------------------------------------------------------------------
__global__ void k_zero1(float* p) { *p = 0.f; }

// Per-row log-softmax NLL, mean-accumulated into *loss via atomics.
__global__ __launch_bounds__(256) void k_loss(const float* __restrict__ logits,
                                              const int* __restrict__ tgt,
                                              float* __restrict__ loss) {
  __shared__ float red[8];
  int row = blockIdx.x;
  const float* lr = logits + (size_t)row * VV;
  int lane = threadIdx.x & 31, wv = threadIdx.x >> 5;
  float mx = -1e30f;
  for (int i = threadIdx.x; i < VV; i += 256) mx = fmaxf(mx, lr[i]);
#pragma unroll
  for (int off = 16; off; off >>= 1) mx = fmaxf(mx, __shfl_xor(mx, off, 32));
  if (!lane) red[wv] = mx;
  __syncthreads();
  if (threadIdx.x == 0) {
    float m = red[0];
    for (int i = 1; i < 8; i++) m = fmaxf(m, red[i]);
    red[0] = m;
  }
  __syncthreads();
  mx = red[0];
  __syncthreads();
  float sum = 0.f;
  for (int i = threadIdx.x; i < VV; i += 256) sum += __expf(lr[i] - mx);
#pragma unroll
  for (int off = 16; off; off >>= 1) sum += __shfl_xor(sum, off, 32);
  if (!lane) red[wv] = sum;
  __syncthreads();
  if (threadIdx.x == 0) {
    float s = 0.f;
    for (int i = 0; i < 8; i++) s += red[i];
    float lse = mx + logf(s);
    float nll = lse - lr[tgt[row]];
    atomicAdd(loss, nll * (1.0f / NROW));
  }
}

// ---------------------------------------------------------------------------
extern "C" void kernel_launch(void* const* d_in, const int* in_sizes, int n_in,
                              void* d_out, int out_size, void* d_ws, size_t ws_size,
                              hipStream_t stream) {
  (void)in_sizes; (void)n_in; (void)out_size; (void)ws_size;
  const int*   idx    = (const int*)d_in[0];
  const int*   target = (const int*)d_in[1];
  const float* tok    = (const float*)d_in[2];
  const float* pos    = (const float*)d_in[3];
  const float* wq     = (const float*)d_in[4];
  const float* wk     = (const float*)d_in[5];
  const float* wvw    = (const float*)d_in[6];
  const float* wproj  = (const float*)d_in[7];
  const float* bproj  = (const float*)d_in[8];
  const float* ln1_g  = (const float*)d_in[9];
  const float* ln1_b  = (const float*)d_in[10];
  const float* ln2_g  = (const float*)d_in[11];
  const float* ln2_b  = (const float*)d_in[12];
  const float* w1     = (const float*)d_in[13];
  const float* b1     = (const float*)d_in[14];
  const float* w2     = (const float*)d_in[15];
  const float* b2     = (const float*)d_in[16];
  const float* lnf_g  = (const float*)d_in[17];
  const float* lnf_b  = (const float*)d_in[18];
  const float* wlm    = (const float*)d_in[19];
  const float* blm    = (const float*)d_in[20];

  char* ws = (char*)d_ws;
  float*    x    = (float*)ws;     ws += (size_t)NROW * DD * 4;
  _Float16* xn   = (_Float16*)ws;  ws += (size_t)NROW * DD * 2;
  _Float16* qb   = (_Float16*)ws;  ws += (size_t)NROW * DD * 2;
  _Float16* kb   = (_Float16*)ws;  ws += (size_t)NROW * DD * 2;
  _Float16* vb   = (_Float16*)ws;  ws += (size_t)NROW * DD * 2;
  _Float16* attn = (_Float16*)ws;  ws += (size_t)NROW * DD * 2;
  _Float16* hb   = (_Float16*)ws;  ws += (size_t)NROW * FFD * 2;
  _Float16* wbuf = (_Float16*)ws;  // 62.5 MB region, reused across stages
  _Float16* wq16 = wbuf;
  _Float16* wk16 = wbuf + (size_t)DD * DD;
  _Float16* wv16 = wbuf + (size_t)2 * DD * DD;
  _Float16* wp16 = wbuf;
  _Float16* w116 = wbuf;
  _Float16* w216 = wbuf + (size_t)DD * FFD;
  _Float16* wl16 = wbuf;

  float* logits = (float*)d_out;
  float* loss   = logits + (size_t)NROW * VV;

  dim3 gD (DD  / BN, NROW / BM);   // N=1024
  dim3 gF (FFD / BN, NROW / BM);   // N=4096
  dim3 gV (VV  / BN, NROW / BM);   // N=32000
  dim3 gAt(TT / QT, HH, BBA);

  k_embed<<<NROW, 256, 0, stream>>>(idx, tok, pos, x);

  for (int l = 0; l < 3; l++) {
    k_layernorm<<<NROW, 256, 0, stream>>>(x, ln1_g + l * DD, ln1_b + l * DD, xn);

    k_repack_hde<<<DD * DD / 1024, 256, 0, stream>>>(wq  + (size_t)l * DD * DD, wq16);
    k_repack_hde<<<DD * DD / 1024, 256, 0, stream>>>(wk  + (size_t)l * DD * DD, wk16);
    k_repack_hde<<<DD * DD / 1024, 256, 0, stream>>>(wvw + (size_t)l * DD * DD, wv16);

    k_gemm<false, false, false, false, true><<<gD, 256, 0, stream>>>(
        xn, wq16, nullptr, nullptr, nullptr, qb, NROW, DD, DD);
    k_gemm<false, false, false, false, true><<<gD, 256, 0, stream>>>(
        xn, wk16, nullptr, nullptr, nullptr, kb, NROW, DD, DD);
    k_gemm<false, false, false, false, true><<<gD, 256, 0, stream>>>(
        xn, wv16, nullptr, nullptr, nullptr, vb, NROW, DD, DD);

    k_attn<<<gAt, 256, ATT_LDS, stream>>>(qb, kb, vb, attn);

    k_f32_to_f16_v4<<<DD * DD / 1024, 256, 0, stream>>>(wproj + (size_t)l * DD * DD, wp16,
                                                        (long long)DD * DD / 4);
    // x = x + attn @ wproj + bproj   (read-modify-write of x is per-element safe)
    k_gemm<true, true, false, true, false><<<gD, 256, 0, stream>>>(
        attn, wp16, bproj + l * DD, x, x, nullptr, NROW, DD, DD);

    k_layernorm<<<NROW, 256, 0, stream>>>(x, ln2_g + l * DD, ln2_b + l * DD, xn);

    k_f32_to_f16_v4<<<DD * FFD / 1024, 256, 0, stream>>>(w1 + (size_t)l * DD * FFD, w116,
                                                         (long long)DD * FFD / 4);
    k_gemm<true, false, true, false, true><<<gF, 256, 0, stream>>>(
        xn, w116, b1 + (size_t)l * FFD, nullptr, nullptr, hb, NROW, FFD, DD);

    k_f32_to_f16_v4<<<FFD * DD / 1024, 256, 0, stream>>>(w2 + (size_t)l * FFD * DD, w216,
                                                         (long long)FFD * DD / 4);
    k_gemm<true, true, false, true, false><<<gD, 256, 0, stream>>>(
        hb, w216, b2 + l * DD, x, x, nullptr, NROW, DD, FFD);
  }

  k_layernorm<<<NROW, 256, 0, stream>>>(x, lnf_g, lnf_b, xn);

  k_f32_to_f16_v4<<<(int)((long long)DD * VV / 1024), 256, 0, stream>>>(
      wlm, wl16, (long long)DD * VV / 4);
  k_gemm<true, false, false, true, false><<<gV, 256, 0, stream>>>(
      xn, wl16, blm, nullptr, logits, nullptr, NROW, VV, DD);

  k_zero1<<<1, 1, 0, stream>>>(loss);
  k_loss<<<NROW, 256, 0, stream>>>(logits, target, loss);
}
